// CrossAttn_46797963657494
// MI455X (gfx1250) — compile-verified, matrix operand order checked by reading, not run on hardware
//
#include <hip/hip_runtime.h>
#include <hip/hip_bf16.h>

// ---------------------------------------------------------------------------
// Deformable cross-attention, MI455X (gfx1250) WMMA implementation.
// B=8, C=256, H=W=64, Nq=4096, NH=4, NP=4, d=64.
// ---------------------------------------------------------------------------

typedef __attribute__((ext_vector_type(16))) _Float16 v16h;
typedef __attribute__((ext_vector_type(8)))  _Float16 v8h;
typedef __attribute__((ext_vector_type(8)))  float    v8f;

#define B_   8
#define C_   256
#define HH   64
#define WW   64
#define NQ   4096
#define NH_  4
#define NP_  4
#define D_   64
#define ROWS (B_ * NQ)          // 32768

// ---- workspace layout (bytes) --------------------------------------------
#define OFF_WTVAL   0ull                                    // 256*256 f16
#define OFF_WTOUT   (OFF_WTVAL + 131072ull)                 // 256*256 f16
#define OFF_WTOA    (OFF_WTOUT + 131072ull)                 // 48*256 f16
#define OFF_QLN     (OFF_WTOA  + 24576ull)                  // ROWS*256 f16
#define OFF_VLN     (OFF_QLN   + 16777216ull)               // ROWS*256 f16
#define OFF_VALUE   (OFF_VLN   + 16777216ull)               // B*NH*NQ*64 f32
#define OFF_OFFATT  (OFF_VALUE + 33554432ull)               // ROWS*48 f32
#define OFF_SAMPLED (OFF_OFFATT + 6291456ull)               // ROWS*256 f16

// ---------------------------------------------------------------------------
// WMMA fragment loader: 16-bit 16x32 A (or symmetric B) fragment from a
// row-major f16 matrix. lane L: row m = L&15, half = L>>4.
// v16h element e (VGPR e/2, slot e%2):
//   e<8 : K = 8*half + e        e>=8: K = 16 + 8*half + (e-8)
// => two contiguous 16B loads per lane.
// ---------------------------------------------------------------------------
__device__ inline v16h load_frag(const _Float16* base, int ld, int lane) {
    const int m = lane & 15, half = lane >> 4;
    const _Float16* p = base + (size_t)m * ld + half * 8;
    v8h lo = *(const v8h*)p;
    v8h hi = *(const v8h*)(p + 16);
    v16h f;
#pragma unroll
    for (int i = 0; i < 8; ++i) { f[i] = lo[i]; f[i + 8] = hi[i]; }
    return f;
}

__device__ inline v8f zero_v8f() {
    v8f z;
#pragma unroll
    for (int i = 0; i < 8; ++i) z[i] = 0.0f;
    return z;
}

// ---------------------------------------------------------------------------
// Kernel 1: weights -> transposed f16 copies.
// Wt_val[n*256+k] = W_val[k*256+n]; Wt_out likewise;
// Wt_oa rows 0..31 = W_off columns, rows 32..47 = W_att columns.
// ---------------------------------------------------------------------------
__global__ __launch_bounds__(256) void prep_weights(
    const float* __restrict__ W_val, const float* __restrict__ W_off,
    const float* __restrict__ W_att, const float* __restrict__ W_out,
    _Float16* __restrict__ Wt_val, _Float16* __restrict__ Wt_out,
    _Float16* __restrict__ Wt_oa) {
    int t = blockIdx.x * 256 + threadIdx.x;      // 65536 threads
    int n = t >> 8, k = t & 255;
    Wt_val[t] = (_Float16)W_val[k * C_ + n];
    Wt_out[t] = (_Float16)W_out[k * C_ + n];
    if (n < 48) {
        float w = (n < 32) ? W_off[k * 32 + n] : W_att[k * 16 + (n - 32)];
        Wt_oa[n * C_ + k] = (_Float16)w;
    }
}

// ---------------------------------------------------------------------------
// Kernel 2/3: LayerNorm over C for a tile of 32 pixels, optional sinusoidal
// positional encoding, output row-major f16 (B*Nq, C).
// Note qpos: inv_freq[j] = 10000^(-j/128), c<128 -> sin, c>=128 -> cos.
// ---------------------------------------------------------------------------
__global__ __launch_bounds__(256) void ln_kernel(
    const float* __restrict__ x,      // (B, C, Nq)
    const float* __restrict__ g, const float* __restrict__ bvec,
    const float* __restrict__ pos_scale,
    _Float16* __restrict__ outp,      // (B*Nq, C) f16
    int addPos) {
    __shared__ float tile[C_][33];    // [channel][query], padded
    __shared__ float red_s[8][32], red_q[8][32];
    __shared__ float mean_s[32], inv_s[32];

    const int t = threadIdx.x;
    const long blk = blockIdx.x;              // 1024 blocks
    const long b  = blk >> 7;                 // 128 blocks per batch
    const long q0 = (blk & 127) * 32;
    const float* xb = x + b * (long)C_ * NQ;

    // coalesced load: 8 channel-rows x 32 query-cols per pass
    const int qc = t & 31, cr = t >> 5;
#pragma unroll
    for (int c0 = 0; c0 < C_; c0 += 8)
        tile[c0 + cr][qc] = xb[(long)(c0 + cr) * NQ + q0 + qc];
    __syncthreads();

    // per-query mean / var: 8 threads per query
    const int qi = t >> 3, part = t & 7;
    float s = 0.0f, ss = 0.0f;
#pragma unroll
    for (int j = 0; j < 32; ++j) {
        float v = tile[part * 32 + j][qi];
        s += v; ss += v * v;
    }
    red_s[part][qi] = s; red_q[part][qi] = ss;
    __syncthreads();
    if (part == 0) {
        float S = 0.0f, Q = 0.0f;
#pragma unroll
        for (int p = 0; p < 8; ++p) { S += red_s[p][qi]; Q += red_q[p][qi]; }
        float m = S * (1.0f / C_);
        float var = Q * (1.0f / C_) - m * m;
        mean_s[qi] = m;
        inv_s[qi] = rsqrtf(var + 1e-5f);
    }
    __syncthreads();

    const float m = mean_s[qi], inv = inv_s[qi];
    const long q = q0 + qi;
    const float ps = pos_scale[0];
    _Float16* orow = outp + (b * NQ + q) * C_;
#pragma unroll
    for (int j = 0; j < 32; ++j) {
        int c = part * 32 + j;
        float v = (tile[c][qi] - m) * inv * g[c] + bvec[c];
        if (addPos) {
            int jj = c & 127;
            float invf = __expf(-(float)jj * (9.210340371976184f / 128.0f));
            float ph = (float)q * invf;
            v += (c < 128 ? __sinf(ph) : __cosf(ph)) * ps;
        }
        orow[c] = (_Float16)v;
    }
}

// ---------------------------------------------------------------------------
// Kernel 4: value = v_ln @ W_val^T(f16) + b_val -> f32 (B, NH, Nq, d).
// Block = 8 waves, covers 32 rows x 256 cols: wave w -> M-tile w>>2,
// N-tiles 4*(w&3)..+3.  K loop: 8 steps of 32, 4 WMMAs per step.
// ---------------------------------------------------------------------------
__global__ __launch_bounds__(256) void gemm_value(
    const _Float16* __restrict__ A,   // (ROWS, 256) f16
    const _Float16* __restrict__ Wt,  // (256, 256) f16, N-major
    const float* __restrict__ bias, float* __restrict__ valout) {
    const int lane = threadIdx.x & 31;
    const int w = threadIdx.x >> 5;
    const long row0 = (long)blockIdx.x * 32 + (long)(w >> 2) * 16;
    const int ntb = (w & 3) * 4;

    v8f acc[4];
#pragma unroll
    for (int i = 0; i < 4; ++i) acc[i] = zero_v8f();

    const _Float16* Arow = A + (size_t)row0 * C_;
#pragma unroll
    for (int k0 = 0; k0 < C_; k0 += 32) {
        if (k0 < C_ - 32) __builtin_prefetch(Arow + k0 + 32, 0, 1);
        v16h af = load_frag(Arow + k0, C_, lane);
#pragma unroll
        for (int i = 0; i < 4; ++i) {
            v16h bf = load_frag(Wt + (size_t)(ntb + i) * 16 * C_ + k0, C_, lane);
            acc[i] = __builtin_amdgcn_wmma_f32_16x16x32_f16(
                false, af, false, bf, (short)0, acc[i], false, false);
        }
    }

    const int n0 = lane & 15, half = lane >> 4;
    const long b = row0 >> 12;
    const long qb = row0 & 4095;
#pragma unroll
    for (int i = 0; i < 4; ++i) {
        int c = (ntb + i) * 16 + n0;
        float bz = bias[c];
        int h = c >> 6, dd = c & 63;
        float* op = valout + (((long)b * NH_ + h) * NQ) * D_ + dd;
#pragma unroll
        for (int r = 0; r < 8; ++r) {
            long q = qb + r + 8 * half;
            op[q * D_] = acc[i][r] + bz;
        }
    }
}

// ---------------------------------------------------------------------------
// Kernel 5: offatt = q_ln @ [W_off|W_att]^T + bias -> f32 (ROWS, 48).
// Block = 8 waves, each its own M-tile (128 rows/block) x 3 N-tiles.
// ---------------------------------------------------------------------------
__global__ __launch_bounds__(256) void gemm_offatt(
    const _Float16* __restrict__ A, const _Float16* __restrict__ Wt,
    const float* __restrict__ b_off, const float* __restrict__ b_att,
    float* __restrict__ offatt) {
    const int lane = threadIdx.x & 31;
    const int w = threadIdx.x >> 5;
    const long row0 = (long)blockIdx.x * 128 + (long)w * 16;

    v8f acc[3];
#pragma unroll
    for (int i = 0; i < 3; ++i) acc[i] = zero_v8f();

    const _Float16* Arow = A + (size_t)row0 * C_;
#pragma unroll
    for (int k0 = 0; k0 < C_; k0 += 32) {
        if (k0 < C_ - 32) __builtin_prefetch(Arow + k0 + 32, 0, 1);
        v16h af = load_frag(Arow + k0, C_, lane);
#pragma unroll
        for (int i = 0; i < 3; ++i) {
            v16h bf = load_frag(Wt + (size_t)i * 16 * C_ + k0, C_, lane);
            acc[i] = __builtin_amdgcn_wmma_f32_16x16x32_f16(
                false, af, false, bf, (short)0, acc[i], false, false);
        }
    }

    const int n0 = lane & 15, half = lane >> 4;
#pragma unroll
    for (int i = 0; i < 3; ++i) {
        int n = i * 16 + n0;
        float bz = (n < 32) ? b_off[n] : b_att[n - 32];
#pragma unroll
        for (int r = 0; r < 8; ++r)
            offatt[(size_t)(row0 + r + 8 * half) * 48 + n] = acc[i][r] + bz;
    }
}

// ---------------------------------------------------------------------------
// Kernel 6: softmax over NP + bilinear sampling.  One thread per (b,h,q).
// px = x + off_x, py = y + off_y exactly (normalizations cancel).
// value rows are contiguous d=64 f32 -> float4 streaming accumulate.
// ---------------------------------------------------------------------------
__device__ inline void accum_corner(float4* acc, const float* vbase,
                                    int xx, int yy, float wgt) {
    if (xx < 0 || xx >= WW || yy < 0 || yy >= HH) return;
    const float4* vp = (const float4*)(vbase + ((size_t)(yy * WW + xx)) * D_);
#pragma unroll
    for (int i = 0; i < 16; ++i) {
        float4 v = vp[i];
        acc[i].x += wgt * v.x; acc[i].y += wgt * v.y;
        acc[i].z += wgt * v.z; acc[i].w += wgt * v.w;
    }
}

__global__ __launch_bounds__(256) void sample_kernel(
    const float* __restrict__ offatt, const float* __restrict__ value,
    _Float16* __restrict__ sampled) {
    const long idx = (long)blockIdx.x * 256 + threadIdx.x;  // (b,h,q)
    const int q = (int)(idx & 4095);
    const int h = (int)((idx >> 12) & 3);
    const long b = idx >> 14;

    const float* oa = offatt + ((size_t)b * NQ + q) * 48;

    // softmax over NP=4 attention logits for this head
    float lg[NP_];
#pragma unroll
    for (int p = 0; p < NP_; ++p) lg[p] = oa[32 + h * NP_ + p];
    float mx = fmaxf(fmaxf(lg[0], lg[1]), fmaxf(lg[2], lg[3]));
    float e[NP_], se = 0.0f;
#pragma unroll
    for (int p = 0; p < NP_; ++p) { e[p] = __expf(lg[p] - mx); se += e[p]; }
    const float inv_se = 1.0f / se;

    const float x = (float)(q & 63);
    const float y = (float)(q >> 6);
    const float* vbase = value + ((size_t)(b * NH_ + h)) * NQ * D_;

    float4 acc[16];
#pragma unroll
    for (int i = 0; i < 16; ++i) acc[i] = make_float4(0.f, 0.f, 0.f, 0.f);

#pragma unroll
    for (int p = 0; p < NP_; ++p) {
        float ap = e[p] * inv_se;
        float px = x + oa[(h * NP_ + p) * 2 + 0];
        float py = y + oa[(h * NP_ + p) * 2 + 1];
        float fx = floorf(px), fy = floorf(py);
        int x0 = (int)fx, y0 = (int)fy;
        float wx = px - fx, wy = py - fy;
        accum_corner(acc, vbase, x0,     y0,     ap * (1.f - wx) * (1.f - wy));
        accum_corner(acc, vbase, x0 + 1, y0,     ap * wx * (1.f - wy));
        accum_corner(acc, vbase, x0,     y0 + 1, ap * (1.f - wx) * wy);
        accum_corner(acc, vbase, x0 + 1, y0 + 1, ap * wx * wy);
    }

    // pack to f16, layout (B, Nq, NH*d) with c = h*64 + dd
    _Float16* op = sampled + ((size_t)b * NQ + q) * C_ + h * D_;
    const float* af = (const float*)acc;
#pragma unroll
    for (int g8 = 0; g8 < 8; ++g8) {
        union { v8h v; _Float16 hx[8]; } u;
#pragma unroll
        for (int j = 0; j < 8; ++j) u.hx[j] = (_Float16)af[g8 * 8 + j];
        *(v8h*)(op + g8 * 8) = u.v;
    }
}

// ---------------------------------------------------------------------------
// Kernel 7: out = sampled @ W_out^T + b_out + x2, stored (B, C, Nq) f32.
// ---------------------------------------------------------------------------
__global__ __launch_bounds__(256) void gemm_out(
    const _Float16* __restrict__ A, const _Float16* __restrict__ Wt,
    const float* __restrict__ bias, const float* __restrict__ x2,
    float* __restrict__ outp) {
    const int lane = threadIdx.x & 31;
    const int w = threadIdx.x >> 5;
    const long row0 = (long)blockIdx.x * 32 + (long)(w >> 2) * 16;
    const int ntb = (w & 3) * 4;

    v8f acc[4];
#pragma unroll
    for (int i = 0; i < 4; ++i) acc[i] = zero_v8f();

    const _Float16* Arow = A + (size_t)row0 * C_;
#pragma unroll
    for (int k0 = 0; k0 < C_; k0 += 32) {
        if (k0 < C_ - 32) __builtin_prefetch(Arow + k0 + 32, 0, 1);
        v16h af = load_frag(Arow + k0, C_, lane);
#pragma unroll
        for (int i = 0; i < 4; ++i) {
            v16h bf = load_frag(Wt + (size_t)(ntb + i) * 16 * C_ + k0, C_, lane);
            acc[i] = __builtin_amdgcn_wmma_f32_16x16x32_f16(
                false, af, false, bf, (short)0, acc[i], false, false);
        }
    }

    const int n0 = lane & 15, half = lane >> 4;
    const long b = row0 >> 12;
    const long qb = row0 & 4095;
#pragma unroll
    for (int i = 0; i < 4; ++i) {
        int c = (ntb + i) * 16 + n0;
        float bz = bias[c];
        const size_t base = ((size_t)b * C_ + c) * NQ + qb;
#pragma unroll
        for (int r = 0; r < 8; ++r) {
            size_t addr = base + r + 8 * half;
            outp[addr] = acc[i][r] + bz + x2[addr];
        }
    }
}

// ---------------------------------------------------------------------------
extern "C" void kernel_launch(void* const* d_in, const int* in_sizes, int n_in,
                              void* d_out, int out_size, void* d_ws, size_t ws_size,
                              hipStream_t stream) {
    const float* x1        = (const float*)d_in[0];
    const float* x2        = (const float*)d_in[1];
    const float* ln1_g     = (const float*)d_in[2];
    const float* ln1_b     = (const float*)d_in[3];
    const float* ln2_g     = (const float*)d_in[4];
    const float* ln2_b     = (const float*)d_in[5];
    const float* pos_scale = (const float*)d_in[6];
    const float* W_off     = (const float*)d_in[7];
    const float* b_off     = (const float*)d_in[8];
    const float* W_att     = (const float*)d_in[9];
    const float* b_att     = (const float*)d_in[10];
    const float* W_val     = (const float*)d_in[11];
    const float* b_val     = (const float*)d_in[12];
    const float* W_out     = (const float*)d_in[13];
    const float* b_out     = (const float*)d_in[14];
    float* outp = (float*)d_out;

    char* ws = (char*)d_ws;
    _Float16* Wt_val  = (_Float16*)(ws + OFF_WTVAL);
    _Float16* Wt_out  = (_Float16*)(ws + OFF_WTOUT);
    _Float16* Wt_oa   = (_Float16*)(ws + OFF_WTOA);
    _Float16* q_ln    = (_Float16*)(ws + OFF_QLN);
    _Float16* v_ln    = (_Float16*)(ws + OFF_VLN);
    float*    value   = (float*)   (ws + OFF_VALUE);
    float*    offatt  = (float*)   (ws + OFF_OFFATT);
    _Float16* sampled = (_Float16*)(ws + OFF_SAMPLED);

    prep_weights<<<256, 256, 0, stream>>>(W_val, W_off, W_att, W_out,
                                          Wt_val, Wt_out, Wt_oa);
    ln_kernel<<<ROWS / 32, 256, 0, stream>>>(x1, ln1_g, ln1_b, pos_scale, q_ln, 1);
    ln_kernel<<<ROWS / 32, 256, 0, stream>>>(x2, ln2_g, ln2_b, pos_scale, v_ln, 0);
    gemm_value<<<ROWS / 32, 256, 0, stream>>>(v_ln, Wt_val, b_val, value);
    gemm_offatt<<<ROWS / 128, 256, 0, stream>>>(q_ln, Wt_oa, b_off, b_att, offatt);
    sample_kernel<<<(B_ * NH_ * NQ) / 256, 256, 0, stream>>>(offatt, value, sampled);
    gemm_out<<<ROWS / 32, 256, 0, stream>>>(sampled, Wt_out, b_out, x2, outp);
}
